// PoseNet_7206955123408
// MI455X (gfx1250) — matrix operand order repaired and checked by reading, not compiled, not run on hardware
//
#include <hip/hip_runtime.h>
#include <hip/hip_bf16.h>

// PoseNet: SE-gated concat features -> LSTM(1024) over T=512 -> Dense(6)
// B=64 T=512 DV=768 DI=256 D=1024 H=1024
//
// Strategy (MI455X / gfx1250, wave32, WMMA bf16 16x16x32, f32 accum):
//   K0  convert w_x, w_h to bf16 (L2-resident: w_h bf16 = 8MB << 192MB L2)
//   K1  fused SE gating -> xg bf16 [32768 x 1024]
//   K2  WMMA GEMM: xz = xg @ w_x_bf + b   (M=32768,N=4096,K=1024) f32 out
//       block tile 128x128, wave tile 32x64 -> 8 WMMA per 6 fragment loads
//   K3  512x sequential LSTM step kernels: z = xz[:,t] + h @ w_h_bf (WMMA),
//       gates, c/h update. 64 blocks, each owns a disjoint 16-col h slice
//       (gate cols {g*1024+js..js+15}); h ping-pongs across launches.
//   K4  out = 0.01*(hs @ fc_w + fc_b)
//
// Workspace layout (needs ~756 MB of d_ws).

#define BATCH 64
#define TSTEPS 512
#define DV 768
#define DI 256
#define DIM 1024
#define HID 1024
#define N4H 4096
#define ROWS (BATCH * TSTEPS) // 32768

typedef __bf16 bf16;
typedef bf16  v16bf __attribute__((ext_vector_type(16)));
typedef float v8f   __attribute__((ext_vector_type(8)));

// ---------------- WMMA fragment helpers (ISA 7.12.2 layouts) ----------------

// A fragment: 16x32 bf16, row-major tile in LDS with given half-stride.
// lane m (0..15): row base+m, K = 0..7 then 16..23
// lane m+16     : row base+m, K = 8..15 then 24..31
__device__ __forceinline__ v16bf load_a_frag(const bf16* tile, int row0,
                                             int stride, int lane) {
  int r  = row0 + (lane & 15);
  int kb = (lane < 16) ? 0 : 8;
  const bf16* p = tile + r * stride + kb;
  v16bf f;
#pragma unroll
  for (int e = 0; e < 8; ++e) f[e] = p[e];
#pragma unroll
  for (int e = 0; e < 8; ++e) f[8 + e] = p[16 + e];
  return f;
}

// B fragment: 32x16 bf16 from a TRANSPOSED LDS tile laid out [n][k]
// (stride = padded k-extent in halves).
// lane n (0..15): col n, K = 0..7 then 16..23 ; lane n+16: col n, K = 8..15 then 24..31
__device__ __forceinline__ v16bf load_b_frag_t(const bf16* tileT, int stride,
                                               int lane) {
  int n  = lane & 15;
  int kb = (lane < 16) ? 0 : 8;
  const bf16* p = tileT + n * stride + kb;
  v16bf f;
#pragma unroll
  for (int e = 0; e < 8; ++e) f[e] = p[e];
#pragma unroll
  for (int e = 0; e < 8; ++e) f[8 + e] = p[16 + e];
  return f;
}

__device__ __forceinline__ v8f wmma_bf16(v16bf a, v16bf b, v8f c) {
  return __builtin_amdgcn_wmma_f32_16x16x32_bf16(false, a, false, b,
                                                 (short)0, c, false, false);
}

__device__ __forceinline__ float sigmoidf(float x) {
  return 1.0f / (1.0f + __expf(-x));
}

// ---------------- K0: f32 -> bf16 convert ----------------
__global__ void __launch_bounds__(256) f32_to_bf16_kernel(
    const float* __restrict__ src, bf16* __restrict__ dst, int n) {
  int i = blockIdx.x * 256 + threadIdx.x;
  if (i < n) dst[i] = (bf16)src[i];
}

// ---------------- zero init ----------------
__global__ void __launch_bounds__(256) zero_f32_kernel(float* __restrict__ p,
                                                       int n) {
  int i = blockIdx.x * 256 + threadIdx.x;
  if (i < n) p[i] = 0.0f;
}

// ---------------- K1: SE gating, one row (b*T+t) per block ----------------
__global__ void __launch_bounds__(256) se_gate_kernel(
    const float* __restrict__ vis, const float* __restrict__ imu,
    const float* __restrict__ w1, const float* __restrict__ b1,
    const float* __restrict__ w2, const float* __restrict__ b2,
    bf16* __restrict__ xg) {
  __shared__ float xrow[DIM];
  __shared__ float red[256];
  __shared__ float t1[64];
  int bt = blockIdx.x;
  int tid = threadIdx.x;

  // concat visual|imu into LDS
#pragma unroll
  for (int j = 0; j < 4; ++j) {
    int d = tid + j * 256;
    xrow[d] = (d < DV) ? vis[(size_t)bt * DV + d]
                       : imu[(size_t)bt * DI + (d - DV)];
  }
  __syncthreads();

  // layer 1: 1024 -> 64, 4 threads per output
  {
    int o = tid >> 2, p = tid & 3;
    float s = 0.0f;
    for (int k = p; k < DIM; k += 4) s += xrow[k] * w1[k * 64 + o];
    red[tid] = s;
  }
  __syncthreads();
  if (tid < 64) {
    float v = red[tid * 4] + red[tid * 4 + 1] + red[tid * 4 + 2] +
              red[tid * 4 + 3] + b1[tid];
    t1[tid] = v > 0.0f ? v : 0.0f;
  }
  __syncthreads();

  // layer 2: 64 -> 1024, sigmoid gate, apply, emit bf16
#pragma unroll
  for (int j = 0; j < 4; ++j) {
    int d = tid + j * 256;
    float s = 0.0f;
#pragma unroll 16
    for (int k = 0; k < 64; ++k) s += t1[k] * w2[k * DIM + d];
    float gate = sigmoidf(s + b2[d]);
    xg[(size_t)bt * DIM + d] = (bf16)(xrow[d] * gate);
  }
}

// ---------------- K2: xz = xg @ w_x_bf + bias  (bf16 WMMA, f32 out) --------
// Block tile 128(M) x 128(N), K-step 32. 8 waves arranged 4(M) x 2(N),
// each wave computes 32x64 = 2x4 accumulator tiles -> 8 WMMA per K-step
// against 2 A-fragment + 4 B-fragment LDS loads (compute-bound GEMM:
// maximize WMMA : DS-load ratio).
__global__ void __launch_bounds__(256) gemm_xz_kernel(
    const bf16* __restrict__ A,    // [32768][1024]
    const bf16* __restrict__ Bw,   // [1024][4096]
    const float* __restrict__ bias,// [4096]
    float* __restrict__ C) {       // [32768][4096]
  __shared__ bf16 At[128 * 34];
  __shared__ bf16 Bt[128 * 34];  // transposed [n][k]
  const int tid = threadIdx.x, lane = tid & 31, w = tid >> 5;
  const int bm = blockIdx.y * 128;
  const int bn = blockIdx.x * 128;
  const int wm = (w >> 1) * 32;   // 0,32,64,96
  const int wn = (w & 1) * 64;    // 0,64

  v8f acc[2][4];
#pragma unroll
  for (int i = 0; i < 2; ++i)
#pragma unroll
    for (int j = 0; j < 4; ++j) acc[i][j] = (v8f)0.0f;

  for (int kk = 0; kk < DIM; kk += 32) {
    {  // A tile 128x32: 16 halves per thread, coalesced rows
      int r = tid >> 1, c0 = (tid & 1) * 16;
      const bf16* s = A + (size_t)(bm + r) * DIM + kk + c0;
#pragma unroll
      for (int e = 0; e < 16; ++e) At[r * 34 + c0 + e] = s[e];
    }
    {  // B tile 32x128 -> transposed LDS [n][k], 16 halves per thread
      int k = tid >> 3, c0 = (tid & 7) * 16;
      const bf16* s = Bw + (size_t)(kk + k) * N4H + bn + c0;
#pragma unroll
      for (int e = 0; e < 16; ++e) Bt[(c0 + e) * 34 + k] = s[e];
    }
    __syncthreads();

    v16bf af[2], bfr[4];
    af[0] = load_a_frag(At, wm, 34, lane);
    af[1] = load_a_frag(At, wm + 16, 34, lane);
#pragma unroll
    for (int nt = 0; nt < 4; ++nt)
      bfr[nt] = load_b_frag_t(&Bt[(wn + nt * 16) * 34], 34, lane);
#pragma unroll
    for (int mt = 0; mt < 2; ++mt)
#pragma unroll
      for (int nt = 0; nt < 4; ++nt)
        acc[mt][nt] = wmma_bf16(af[mt], bfr[nt], acc[mt][nt]);
    __syncthreads();
  }

  // epilogue: + bias, fp32 store (C/D layout: lane n=lane&15, rows rb+e)
  int n = lane & 15, rb = (lane < 16) ? 0 : 8;
#pragma unroll
  for (int mt = 0; mt < 2; ++mt)
#pragma unroll
    for (int nt = 0; nt < 4; ++nt) {
      int col = bn + wn + nt * 16 + n;
      float bv = bias[col];
#pragma unroll
      for (int e = 0; e < 8; ++e)
        C[(size_t)(bm + wm + mt * 16 + rb + e) * N4H + col] =
            acc[mt][nt][e] + bv;
    }
}

// ---------------- K3: one LSTM time step --------------------------------
// 64 blocks; block bn owns h-cols js=bn*16..+15 -> gate cols g*1024+js+n.
// 8 waves: wave w -> gate g=w>>1, row-half mh=(w&1)*32; 2 WMMA accs each.
__global__ void __launch_bounds__(256) lstm_step_kernel(
    const float* __restrict__ hprev,  // [64][1024] f32
    float* __restrict__ hnew,         // [64][1024] f32
    float* __restrict__ cbuf,         // [64][1024] f32
    const bf16* __restrict__ Wh,      // [1024][4096] bf16
    const float* __restrict__ xz,     // [32768][4096] f32 (row = b*T+t)
    float* __restrict__ hs,           // [64][512][1024] f32
    int t) {
  __shared__ bf16 At[64 * 34];
  __shared__ bf16 Bt[4 * 16 * 34];          // [g][n][k] transposed
  __shared__ float zbuf[4][64][16];         // [gate][row][col]
  const int tid = threadIdx.x, lane = tid & 31, w = tid >> 5;
  const int js = blockIdx.x * 16;
  const int g = w >> 1;
  const int mh = (w & 1) * 32;

  v8f acc0 = (v8f)0.0f, acc1 = (v8f)0.0f;

  for (int kk = 0; kk < HID; kk += 32) {
    {  // A tile: h_prev 64x32 f32 -> bf16 LDS
      int r = tid >> 2, c0 = (tid & 3) * 8;
      const float* s = hprev + r * HID + kk + c0;
#pragma unroll
      for (int e = 0; e < 8; ++e) At[r * 34 + c0 + e] = (bf16)s[e];
    }
    {  // B tile: w_h rows kk..kk+31, 4x16 gate cols -> transposed LDS
      int k = tid >> 3, cc0 = (tid & 7) * 8;
      int gg = cc0 >> 4, n0 = cc0 & 15;
      const bf16* s = Wh + (size_t)(kk + k) * N4H + gg * HID + js + n0;
#pragma unroll
      for (int e = 0; e < 8; ++e) Bt[(gg * 16 + n0 + e) * 34 + k] = s[e];
    }
    __syncthreads();

    v16bf bfrag = load_b_frag_t(&Bt[(g * 16) * 34], 34, lane);
    v16bf a0 = load_a_frag(At, mh, 34, lane);
    v16bf a1 = load_a_frag(At, mh + 16, 34, lane);
    acc0 = wmma_bf16(a0, bfrag, acc0);
    acc1 = wmma_bf16(a1, bfrag, acc1);
    __syncthreads();
  }

  // spill per-gate z to LDS for cross-wave combine
  {
    int n = lane & 15, rb = (lane < 16) ? 0 : 8;
#pragma unroll
    for (int e = 0; e < 8; ++e) {
      zbuf[g][mh + rb + e][n] = acc0[e];
      zbuf[g][mh + 16 + rb + e][n] = acc1[e];
    }
  }
  __syncthreads();

  // gate combine + state update: 1024 states / 256 threads = 4 each
#pragma unroll
  for (int it = 0; it < 4; ++it) {
    int idx = tid + it * 256;
    int row = idx >> 4, col = idx & 15;            // row = batch index
    size_t xzb = ((size_t)row * TSTEPS + t) * N4H + js + col;
    float zi = zbuf[0][row][col] + xz[xzb + 0 * HID];
    float zf = zbuf[1][row][col] + xz[xzb + 1 * HID];
    float zg = zbuf[2][row][col] + xz[xzb + 2 * HID];
    float zo = zbuf[3][row][col] + xz[xzb + 3 * HID];
    float ig = sigmoidf(zi);
    float fg = sigmoidf(zf);
    float gg = tanhf(zg);
    float og = sigmoidf(zo);
    int hidx = row * HID + js + col;               // slice owned by block
    float c = fg * cbuf[hidx] + ig * gg;
    cbuf[hidx] = c;
    float h = og * tanhf(c);
    hnew[hidx] = h;
    hs[((size_t)row * TSTEPS + t) * HID + js + col] = h;
  }
}

// ---------------- K4: out = 0.01*(hs @ fc_w + fc_b) ----------------
__global__ void __launch_bounds__(256) final_dense_kernel(
    const float* __restrict__ hs, const float* __restrict__ fw,
    const float* __restrict__ fb, float* __restrict__ out) {
  int tid = threadIdx.x;
  int row = blockIdx.x * 32 + (tid >> 3);
  int c = tid & 7;
  if (c >= 6) return;
  const float* h = hs + (size_t)row * HID;
  float s = 0.0f;
  for (int k = 0; k < HID; ++k) s += h[k] * fw[k * 6 + c];
  out[(size_t)row * 6 + c] = 0.01f * (s + fb[c]);
}

// ---------------- host side ----------------
extern "C" void kernel_launch(void* const* d_in, const int* in_sizes, int n_in,
                              void* d_out, int out_size, void* d_ws,
                              size_t ws_size, hipStream_t stream) {
  const float* vis   = (const float*)d_in[0];
  const float* imu   = (const float*)d_in[1];
  const float* se_w1 = (const float*)d_in[2];
  const float* se_b1 = (const float*)d_in[3];
  const float* se_w2 = (const float*)d_in[4];
  const float* se_b2 = (const float*)d_in[5];
  const float* w_x   = (const float*)d_in[6];
  const float* w_h   = (const float*)d_in[7];
  const float* bias  = (const float*)d_in[8];
  const float* fc_w  = (const float*)d_in[9];
  const float* fc_b  = (const float*)d_in[10];
  float* out = (float*)d_out;

  char* ws = (char*)d_ws;
  size_t off = 0;
  auto carve = [&](size_t bytes) -> char* {
    char* p = ws + off;
    off += (bytes + 255) & ~(size_t)255;
    return p;
  };
  const size_t NW = (size_t)DIM * N4H;                  // 4,194,304
  bf16*  wx_bf  = (bf16*)carve(NW * 2);                 //   8 MB
  bf16*  wh_bf  = (bf16*)carve(NW * 2);                 //   8 MB
  bf16*  xg     = (bf16*)carve((size_t)ROWS * DIM * 2); //  64 MB
  float* xz     = (float*)carve((size_t)ROWS * N4H * 4);// 512 MB
  float* hs     = (float*)carve((size_t)ROWS * HID * 4);// 128 MB
  float* hstate = (float*)carve((size_t)3 * BATCH * HID * 4); // hpp0|cbuf|hpp1
  float* hpp0 = hstate;
  float* cbuf = hstate + BATCH * HID;
  float* hpp1 = hstate + 2 * BATCH * HID;

  // K0: weights -> bf16
  f32_to_bf16_kernel<<<(int)((NW + 255) / 256), 256, 0, stream>>>(w_x, wx_bf,
                                                                  (int)NW);
  f32_to_bf16_kernel<<<(int)((NW + 255) / 256), 256, 0, stream>>>(w_h, wh_bf,
                                                                  (int)NW);
  // zero h0 and c0 (hpp0 + cbuf are contiguous)
  zero_f32_kernel<<<(2 * BATCH * HID + 255) / 256, 256, 0, stream>>>(
      hpp0, 2 * BATCH * HID);

  // K1: SE gating
  se_gate_kernel<<<ROWS, 256, 0, stream>>>(vis, imu, se_w1, se_b1, se_w2,
                                           se_b2, xg);

  // K2: xz = xg @ w_x + b
  gemm_xz_kernel<<<dim3(N4H / 128, ROWS / 128), 256, 0, stream>>>(xg, wx_bf,
                                                                  bias, xz);

  // K3: sequential LSTM steps (ping-pong h)
  for (int t = 0; t < TSTEPS; ++t) {
    const float* hp = (t & 1) ? hpp1 : hpp0;
    float* hn = (t & 1) ? hpp0 : hpp1;
    lstm_step_kernel<<<HID / 16, 256, 0, stream>>>(hp, hn, cbuf, wh_bf, xz,
                                                   hs, t);
  }

  // K4: dense head
  final_dense_kernel<<<ROWS / 32, 256, 0, stream>>>(hs, fc_w, fc_b, out);
}
